// MultiheadAttention_21354577395747
// MI455X (gfx1250) — matrix operand order repaired and checked by reading, not compiled, not run on hardware
//
#include <hip/hip_runtime.h>
#include <math.h>

// ---------------- problem constants (from reference setup_inputs) ------------
constexpr int C_   = 32;          // channels
constexpr int H_   = 224;
constexpr int W_   = 336;
constexpr int TQ   = 2;           // frames (tq == t == 2)
constexpr int NH   = 4;           // heads
constexpr int P_   = 8;           // patch
constexpr int BH   = H_ / P_;     // 28
constexpr int BW   = W_ / P_;     // 42
constexpr int NPF  = BH * BW;     // 1176 patches per frame
constexpr int TN   = TQ * NPF;    // 2352 (real sequence length)
constexpr int TNP  = 2432;        // padded to 19 * 128 GEMM blocks
constexpr int D_   = C_ * P_ * P_;// 2048 (feature dim)
constexpr int WOUT = W_ * NH;     // 1344 (head concat along w)
constexpr int WT   = W_ / 16;     // 21 pixel strips per image row

typedef __bf16 bf16;
typedef __attribute__((ext_vector_type(16))) __bf16 v16bf;
typedef __attribute__((ext_vector_type(8)))  __bf16 v8bf;
typedef __attribute__((ext_vector_type(8)))  float  v8f;
typedef __attribute__((ext_vector_type(4)))  int    v4i;

#define AS1 __attribute__((address_space(1)))
#define AS3 __attribute__((address_space(3)))
typedef AS1 v4i* gv4i_p;
typedef AS3 v4i* lv4i_p;

// ------------- CDNA5 async global->LDS copy (guarded probe) ------------------
#if __has_builtin(__builtin_amdgcn_global_load_async_to_lds_b128)
#define HAVE_ASYNC_LDS 1
__device__ __forceinline__ void cp16_async(const void* g, void* s) {
    __builtin_amdgcn_global_load_async_to_lds_b128(
        (gv4i_p)(g), (lv4i_p)(s), 0, 0);
}
#elif __has_builtin(__builtin_amdgcn_global_load_lds)
#define HAVE_ASYNC_LDS 1
__device__ __forceinline__ void cp16_async(const void* g, void* s) {
    __builtin_amdgcn_global_load_lds((gv4i_p)(g), (lv4i_p)(s), 16, 0, 0);
}
#endif

__device__ __forceinline__ void async_wait0() {
#if __has_builtin(__builtin_amdgcn_s_wait_asynccnt)
    __builtin_amdgcn_s_wait_asynccnt(0);
#else
    asm volatile("s_wait_asynccnt 0x0" ::: "memory");
#endif
}

__device__ __forceinline__ v16bf pack16(v8bf lo, v8bf hi) {
    v16bf r;
    #pragma unroll
    for (int e = 0; e < 8; ++e) { r[e] = lo[e]; r[8 + e] = hi[e]; }
    return r;
}

// ---------------- workspace zeroing (pad rows of uq/uk/uv must be 0) ---------
__global__ void zero_ws_k(uint4* __restrict__ p, size_t n16) {
    size_t i  = (size_t)blockIdx.x * blockDim.x + threadIdx.x;
    size_t st = (size_t)gridDim.x * blockDim.x;
    uint4 z; z.x = z.y = z.z = z.w = 0u;
    for (; i < n16; i += st) p[i] = z;
}

// ---------------- NCHW f32 -> NHWC bf16 (value; head-independent) ------------
__global__ __launch_bounds__(256)
void nhwc_k(const float* __restrict__ x, bf16* __restrict__ xh) {
    int idx = blockIdx.x * blockDim.x + threadIdx.x;    // over TQ*H*W pixels
    if (idx >= TQ * H_ * W_) return;
    int w = idx % W_;
    int h = (idx / W_) % H_;
    int f = idx / (H_ * W_);
    const float* xp = x + ((size_t)f * C_ * H_ + h) * W_ + w;
    bf16 tmp[C_];
    #pragma unroll
    for (int ci = 0; ci < C_; ++ci) tmp[ci] = (bf16)xp[(size_t)ci * H_ * W_];
    uint4* o4 = (uint4*)(xh + (size_t)idx * C_);
    const uint4* t4 = (const uint4*)tmp;
    o4[0] = t4[0]; o4[1] = t4[1]; o4[2] = t4[2]; o4[3] = t4[3];
}

// ---------------- depthwise 3x3 conv + unfold -> bf16 [col][d] ---------------
__global__ __launch_bounds__(256)
void dwconv_unfold_k(const float* __restrict__ x,   // [TQ][C][H][W]
                     const float* __restrict__ w9,  // [C][9] (this head)
                     const float* __restrict__ bias,// [C]
                     bf16* __restrict__ u)          // [TNP][D_]
{
    __shared__ float ws[C_ * 9];
    __shared__ float bs[C_];
    int t = threadIdx.x;
    if (t < C_ * 9) ws[t] = w9[t];
    if (t < C_)     bs[t] = bias[t];
    __syncthreads();

    int idx = blockIdx.x * blockDim.x + t;            // over TQ*C*H*W
    if (idx >= TQ * C_ * H_ * W_) return;
    int w  = idx % W_;
    int h  = (idx / W_) % H_;
    int ch = (idx / (W_ * H_)) % C_;
    int f  = idx / (W_ * H_ * C_);

    const float* xp = x + ((size_t)(f * C_ + ch)) * H_ * W_;
    float acc = bs[ch];
    #pragma unroll
    for (int dy = -1; dy <= 1; ++dy) {
        int hh = h + dy;
        bool hv = (hh >= 0) && (hh < H_);
        #pragma unroll
        for (int dx = -1; dx <= 1; ++dx) {
            int ww = w + dx;
            float xv = (hv && ww >= 0 && ww < W_) ? xp[hh * W_ + ww] : 0.0f;
            acc = fmaf(xv, ws[ch * 9 + (dy + 1) * 3 + (dx + 1)], acc);
        }
    }
    int col = f * NPF + (h / P_) * BW + (w / P_);
    int d   = ch * 64 + (h % P_) * P_ + (w % P_);
    u[(size_t)col * D_ + d] = (bf16)acc;
}

// ------- dense 3x3 conv (32->32) as implicit-GEMM WMMA + unfold --------------
// Each wave: one 16-pixel strip (fixed f,h), all 32 out channels.
// A (M=16 pixels, K=32 in-ch) read per-lane from NHWC; B (K=ci, N=co) from LDS.
__global__ __launch_bounds__(128)
void dconv_wmma_k(const bf16* __restrict__ xh,   // [TQ][H][W][C] bf16
                  const float* __restrict__ wv,  // [32][32][9] (this head)
                  const float* __restrict__ bias,// [32]
                  bf16* __restrict__ uv)         // [D_][TNP]
{
    __shared__ bf16 Ws[9 * 32 * 40];               // [tap][co][ci] pad 40
    __shared__ float bs[C_];
    const int tid = threadIdx.x;
    for (int i = tid; i < 9 * 32 * 32; i += 128) {
        int tap = i >> 10, co = (i >> 5) & 31, ci = i & 31;
        Ws[(tap * 32 + co) * 40 + ci] = (bf16)wv[(co * 32 + ci) * 9 + tap];
    }
    if (tid < C_) bs[tid] = bias[tid];
    __syncthreads();

    const int lane = tid & 31, wave = tid >> 5;
    const int lrow = lane & 15, g = lane >> 4;
    const int strip = blockIdx.x * 4 + wave;       // [0, TQ*H*WT)
    const int w0 = (strip % WT) * 16;
    const int h  = (strip / WT) % H_;
    const int f  = strip / (WT * H_);

    v8f acc[2];
    #pragma unroll
    for (int j = 0; j < 2; ++j)
        #pragma unroll
        for (int e = 0; e < 8; ++e) acc[j][e] = 0.0f;

    const size_t frame_off = (size_t)f * H_ * W_ * C_;
    #pragma unroll
    for (int dy = -1; dy <= 1; ++dy) {
        const int hh = h + dy;
        const bool hv = (hh >= 0) && (hh < H_);
        #pragma unroll
        for (int dx = -1; dx <= 1; ++dx) {
            const int tap = (dy + 1) * 3 + (dx + 1);
            const int ww = w0 + lrow + dx;
            const bool valid = hv && (ww >= 0) && (ww < W_);
            const bf16* pa = xh + frame_off +
                             ((size_t)(valid ? hh : 0) * W_ + (valid ? ww : 0)) * C_;
            v8bf lo = *(const v8bf*)(pa + 8 * g);
            v8bf hi = *(const v8bf*)(pa + 16 + 8 * g);
            if (!valid) {
                #pragma unroll
                for (int e = 0; e < 8; ++e) { lo[e] = (bf16)0.0f; hi[e] = (bf16)0.0f; }
            }
            v16bf af = pack16(lo, hi);
            #pragma unroll
            for (int j = 0; j < 2; ++j) {
                const bf16* pb = &Ws[(tap * 32 + j * 16 + lrow) * 40];
                v16bf bfq = pack16(*(const v8bf*)(pb + 8 * g),
                                   *(const v8bf*)(pb + 16 + 8 * g));
                acc[j] = __builtin_amdgcn_wmma_f32_16x16x32_bf16(
                    false, af, false, bfq, (short)0, acc[j], false, false);
            }
        }
    }

    const int ki = h & 7;
    const int col_base = f * NPF + (h >> 3) * BW;
    #pragma unroll
    for (int j = 0; j < 2; ++j) {
        const int co = j * 16 + lrow;
        const float b = bs[co];
        #pragma unroll
        for (int r = 0; r < 8; ++r) {
            const int wpix = w0 + r + 8 * g;
            const int col  = col_base + (wpix >> 3);
            const int d    = co * 64 + ki * 8 + (wpix & 7);
            uv[(size_t)d * TNP + col] = (bf16)(acc[j][r] + b);
        }
    }
}

// ---------------- shared GEMM pieces -----------------------------------------
// 128 threads = 4 waves; block tile 128x128; wave tile 64x64 (4x4 WMMA accs).
// Double-buffered LDS (stride 40 bf16 = 80B, conflict-free fragment reads).
// Global->LDS via async-to-LDS when available (ASYNCcnt), else sync copies.

#if defined(HAVE_ASYNC_LDS)
#define CP64(s_, g_)                                                           \
    { _Pragma("unroll") for (int q_ = 0; q_ < 4; ++q_)                         \
        cp16_async((const char*)(g_) + 16 * q_, (char*)(s_) + 16 * q_); }
#define STAGE_FENCE() do { async_wait0(); __syncthreads(); } while (0)
#else
#define CP64(s_, g_)                                                           \
    { const uint4* g4_ = (const uint4*)(g_); uint4* s4_ = (uint4*)(s_);        \
      s4_[0] = g4_[0]; s4_[1] = g4_[1]; s4_[2] = g4_[2]; s4_[3] = g4_[3]; }
#define STAGE_FENCE() __syncthreads()
#endif

#define GEMM_PROLOG()                                                          \
    __shared__ bf16 As[2][128 * 40];                                           \
    __shared__ bf16 Bs[2][128 * 40];                                           \
    const int tid  = threadIdx.x;      /* 0..127 */                            \
    const int lane = tid & 31;                                                 \
    const int wave = tid >> 5;         /* 0..3 */                              \
    const int wm   = wave >> 1;                                                \
    const int wn   = wave & 1;                                                 \
    const int lrow = lane & 15;                                                \
    const int g    = lane >> 4;                                                \
    const int m0   = blockIdx.y * 128;                                         \
    const int n0   = blockIdx.x * 128;                                         \
    v8f acc[4][4];                                                             \
    _Pragma("unroll") for (int i = 0; i < 4; ++i)                              \
        _Pragma("unroll") for (int j = 0; j < 4; ++j)                          \
            _Pragma("unroll") for (int e = 0; e < 8; ++e) acc[i][j][e] = 0.0f;

#define GEMM_LOAD_STAGE(buf_, k0_, Aptr, Bptr, lda, ldb)                       \
    {                                                                          \
        const bf16* gA_ = (Aptr) + (size_t)(m0 + tid) * (lda) + (k0_);         \
        const bf16* gB_ = (Bptr) + (size_t)(n0 + tid) * (ldb) + (k0_);         \
        CP64(&As[buf_][tid * 40], gA_);                                        \
        CP64(&Bs[buf_][tid * 40], gB_);                                        \
    }

#define GEMM_COMPUTE_STAGE(buf_)                                               \
    {                                                                          \
        v16bf af_[4], bf_[4];                                                  \
        _Pragma("unroll") for (int i = 0; i < 4; ++i) {                        \
            const bf16* pa_ = &As[buf_][(wm * 64 + i * 16 + lrow) * 40];       \
            af_[i] = pack16(*(const v8bf*)(pa_ + 8 * g),                       \
                            *(const v8bf*)(pa_ + 16 + 8 * g));                 \
        }                                                                      \
        _Pragma("unroll") for (int j = 0; j < 4; ++j) {                        \
            const bf16* pb_ = &Bs[buf_][(wn * 64 + j * 16 + lrow) * 40];       \
            bf_[j] = pack16(*(const v8bf*)(pb_ + 8 * g),                       \
                            *(const v8bf*)(pb_ + 16 + 8 * g));                 \
        }                                                                      \
        _Pragma("unroll") for (int i = 0; i < 4; ++i)                          \
            _Pragma("unroll") for (int j = 0; j < 4; ++j)                      \
                acc[i][j] = __builtin_amdgcn_wmma_f32_16x16x32_bf16(           \
                    false, af_[i], false, bf_[j], (short)0, acc[i][j],         \
                    false, false);                                             \
    }

#define GEMM_MAINLOOP(Aptr, Bptr, Kdim, lda, ldb)                              \
    GEMM_LOAD_STAGE(0, 0, Aptr, Bptr, lda, ldb);                               \
    STAGE_FENCE();                                                             \
    {                                                                          \
        const int nk_ = (Kdim) / 32;                                           \
        for (int it_ = 0; it_ < nk_; ++it_) {                                  \
            if (it_ + 1 < nk_)                                                 \
                GEMM_LOAD_STAGE((it_ + 1) & 1, (it_ + 1) * 32,                 \
                                Aptr, Bptr, lda, ldb);                         \
            GEMM_COMPUTE_STAGE(it_ & 1);                                       \
            STAGE_FENCE();                                                     \
        }                                                                      \
    }

// ---------------- GEMM 1: attn = (uq^T uk) * (30 / sqrt(d)) ------------------
__global__ __launch_bounds__(128)
void gemm_qk_k(const bf16* __restrict__ A,    // uq [TNP][D_]
               const bf16* __restrict__ Bt,   // uk [TNP][D_]
               float* __restrict__ Cmat,      // attn [TNP][TNP]
               float scale)
{
    GEMM_PROLOG();
    GEMM_MAINLOOP(A, Bt, D_, D_, D_);
    #pragma unroll
    for (int i = 0; i < 4; ++i) {
        int m = m0 + wm * 64 + i * 16 + g * 8;
        #pragma unroll
        for (int j = 0; j < 4; ++j) {
            int n = n0 + wn * 64 + j * 16 + lrow;
            float* cp = Cmat + (size_t)m * TNP + n;
            #pragma unroll
            for (int r = 0; r < 8; ++r)
                cp[(size_t)r * TNP] = acc[i][j][r] * scale;
        }
    }
}

// ---------------- row softmax (f32 in, bf16 out, pads zeroed) ----------------
__global__ __launch_bounds__(256)
void softmax_row_k(const float* __restrict__ attn, bf16* __restrict__ asmx)
{
    const int q = blockIdx.x;
    const float* row = attn + (size_t)q * TNP;
    __shared__ float rowc[TN];
    __shared__ float red[256];
    const int t = threadIdx.x;

    float mx = -3.0e38f;
    for (int k = t; k < TN; k += 256) {
        float v = row[k];
        rowc[k] = v;
        mx = fmaxf(mx, v);
    }
    red[t] = mx; __syncthreads();
    #pragma unroll
    for (int s = 128; s > 0; s >>= 1) {
        if (t < s) red[t] = fmaxf(red[t], red[t + s]);
        __syncthreads();
    }
    mx = red[0]; __syncthreads();

    float sum = 0.0f;
    for (int k = t; k < TN; k += 256) sum += __expf(rowc[k] - mx);
    red[t] = sum; __syncthreads();
    #pragma unroll
    for (int s = 128; s > 0; s >>= 1) {
        if (t < s) red[t] += red[t + s];
        __syncthreads();
    }
    const float inv = 1.0f / red[0];

    bf16* orow = asmx + (size_t)q * TNP;
    for (int k = t; k < TNP; k += 256)
        orow[k] = (k < TN) ? (bf16)(__expf(rowc[k] - mx) * inv) : (bf16)0.0f;
}

// ---------------- GEMM 2: out = attn_sm * uv^T, fused fold + head concat -----
__global__ __launch_bounds__(128)
void gemm_av_k(const bf16* __restrict__ A,    // attn_sm [TNP][TNP]
               const bf16* __restrict__ Bt,   // uv [D_][TNP]
               float* __restrict__ out,       // [TQ][C][H][W*NH]
               int head)
{
    GEMM_PROLOG();
    GEMM_MAINLOOP(A, Bt, TNP, TNP, TNP);
    #pragma unroll
    for (int i = 0; i < 4; ++i) {
        int mbase = m0 + wm * 64 + i * 16 + g * 8;
        #pragma unroll
        for (int j = 0; j < 4; ++j) {
            int d = n0 + wn * 64 + j * 16 + lrow;    // 0..2047, always valid
            int cidx = d >> 6;
            int ki   = (d & 63) >> 3;
            int kj   = d & 7;
            #pragma unroll
            for (int r = 0; r < 8; ++r) {
                int q = mbase + r;
                if (q < TN) {
                    int f  = q / NPF;
                    int p  = q % NPF;
                    int bi = p / BW;
                    int bj = p % BW;
                    int hh = bi * P_ + ki;
                    int ww = head * W_ + bj * P_ + kj;
                    out[(((size_t)f * C_ + cidx) * H_ + hh) * WOUT + ww] =
                        acc[i][j][r];
                }
            }
        }
    }
}

// ---------------- host orchestration -----------------------------------------
extern "C" void kernel_launch(void* const* d_in, const int* in_sizes, int n_in,
                              void* d_out, int out_size, void* d_ws, size_t ws_size,
                              hipStream_t stream) {
    const float* query = (const float*)d_in[0];
    const float* key   = (const float*)d_in[1];
    const float* value = (const float*)d_in[2];
    const float* Wq    = (const float*)d_in[3];
    const float* bq    = (const float*)d_in[4];
    const float* Wk    = (const float*)d_in[5];
    const float* bk    = (const float*)d_in[6];
    const float* Wv    = (const float*)d_in[7];
    const float* bv    = (const float*)d_in[8];
    float* out = (float*)d_out;

    // workspace layout (per-head buffers reused across heads): ~75 MB
    char* ws = (char*)d_ws;
    const size_t u_bytes    = (size_t)TNP * D_ * sizeof(bf16);    // 9.96 MB
    const size_t attn_bytes = (size_t)TNP * TNP * sizeof(float);  // 23.7 MB
    const size_t asm_bytes  = (size_t)TNP * TNP * sizeof(bf16);   // 11.8 MB
    bf16*  uq     = (bf16*)ws;
    bf16*  uk     = (bf16*)(ws + u_bytes);
    bf16*  uv     = (bf16*)(ws + 2 * u_bytes);
    float* attn   = (float*)(ws + 3 * u_bytes);
    bf16*  attnsm = (bf16*)(ws + 3 * u_bytes + attn_bytes);
    bf16*  xh     = (bf16*)(ws + 3 * u_bytes + attn_bytes + asm_bytes);

    // zero uq/uk/uv once so padded rows (TN..TNP) contribute exact zeros
    size_t n16 = (3 * u_bytes) / 16;
    zero_ws_k<<<2048, 256, 0, stream>>>((uint4*)ws, n16);

    // value -> NHWC bf16 once (head independent)
    nhwc_k<<<(TQ * H_ * W_) / 256, 256, 0, stream>>>(value, xh);

    const float scale = 30.0f / sqrtf((float)D_);     // softmax temp * d^-0.5
    const int dw_blocks = (TQ * C_ * H_ * W_) / 256;  // 18816
    const int dc_blocks = (TQ * H_ * WT) / 4;         // 2352 (4 waves/block)
    dim3 g_qk(TNP / 128, TNP / 128);                  // (19, 19)
    dim3 g_av(D_ / 128, TNP / 128);                   // (16, 19)

    for (int hd = 0; hd < NH; ++hd) {
        dwconv_unfold_k<<<dw_blocks, 256, 0, stream>>>(
            query, Wq + (size_t)hd * C_ * 9, bq + (size_t)hd * C_, uq);
        dwconv_unfold_k<<<dw_blocks, 256, 0, stream>>>(
            key,   Wk + (size_t)hd * C_ * 9, bk + (size_t)hd * C_, uk);
        dconv_wmma_k<<<dc_blocks, 128, 0, stream>>>(
            xh, Wv + (size_t)hd * C_ * C_ * 9, bv + (size_t)hd * C_, uv);
        gemm_qk_k<<<g_qk, 128, 0, stream>>>(uq, uk, attn, scale);
        softmax_row_k<<<TNP, 256, 0, stream>>>(attn, attnsm);
        gemm_av_k<<<g_av, 128, 0, stream>>>(attnsm, uv, out, hd);
    }
}
